// SGCN_35605278884340
// MI455X (gfx1250) — matrix operand order repaired and checked by reading, not compiled, not run on hardware
//
#include <hip/hip_runtime.h>

typedef __attribute__((ext_vector_type(16))) __bf16 v16bf;
typedef __attribute__((ext_vector_type(8)))  float  v8f;

#define NB 16
#define TT 512
#define VV 25
#define CCH 256
#define CO 256
#define KKG 3
#define KO 768   /* K*Co */
#define KT 9
#define EPSV 1e-5f

__device__ __forceinline__ unsigned short f2bf(float f) {
    unsigned u = __float_as_uint(f);
    unsigned r = (u + 0x7FFFu + ((u >> 16) & 1u)) >> 16;   // round-to-nearest-even
    return (unsigned short)r;
}
__device__ __forceinline__ float bf2f(unsigned short h) {
    return __uint_as_float(((unsigned)h) << 16);
}

union Frag { uint4 q[2]; v16bf v; };

// ---------------------------------------------------------------------------
// 1) BatchNorm (eval) + transpose (N,T,V,C)->[n*V+v][t][c] + fp32->bf16
// ---------------------------------------------------------------------------
__global__ void bn_pack_kernel(const float* __restrict__ x,
                               const float* __restrict__ gam,
                               const float* __restrict__ bet,
                               const float* __restrict__ mu,
                               const float* __restrict__ var,
                               unsigned short* __restrict__ xb) {
    long long i = (long long)blockIdx.x * blockDim.x + threadIdx.x; // over N*T*V*C
    int c = (int)(i % CCH);
    int v = (int)((i / CCH) % VV);
    long long rest = i / ((long long)CCH * VV);
    int t = (int)(rest % TT);
    int n = (int)(rest / TT);
    int ch = v * CCH + c;                 // V-major, C-minor channel index
    float inv = gam[ch] * rsqrtf(var[ch] + EPSV);
    float val = (x[i] - mu[ch]) * inv + bet[ch];
    long long o = (((long long)(n * VV + v)) * TT + t) * CCH + c;
    xb[o] = f2bf(val);
}

// ---------------------------------------------------------------------------
// 2) Weight pre-swizzle: w[ko][c][dt] (fp32) -> bf16 tiles in the exact
//    32x16 bf16 B-fragment VGPR order: tile(dt,cc,koT), 32 lanes x 16 elems.
//    lane element e: K = (e<8 ? 0 : 16) + (lane>=16?8:0) + (e&7), N = lane&15.
// ---------------------------------------------------------------------------
__global__ void pack_w_kernel(const float* __restrict__ w,
                              unsigned short* __restrict__ bp) {
    int idx = blockIdx.x * blockDim.x + threadIdx.x;   // 9*8*48*512 total
    int e    = idx & 15;
    int lane = (idx >> 4) & 31;
    int tile = idx >> 9;
    int koT  = tile % 48;
    int cc   = (tile / 48) % 8;
    int dt   = tile / (48 * 8);
    int k  = ((e >> 3) * 16) + ((lane >> 4) * 8) + (e & 7);
    int c  = cc * 32 + k;
    int ko = koT * 16 + (lane & 15);
    bp[idx] = f2bf(w[((long long)ko * CCH + c) * KT + dt]);
}

// helper: per-lane ushort offset into a packed 32x16 B tile
__device__ __forceinline__ int lane_chunk(int row, int hi) {
    return (hi * 16 + row) * 16;   // lane * 16 ushorts
}

// ---------------------------------------------------------------------------
// 3) Temporal-conv GEMM. Wave tile: 32(M) x 64(N), 8 accumulators.
//    CHECK=false is the steady-state path (no halo tests, pure b128 loads).
//    All 4 B fragments are pulled into distinct live registers per step so the
//    scheduler can keep a 4-deep load pipeline (staggered s_wait_loadcnt).
// ---------------------------------------------------------------------------
template <bool CHECK>
__device__ __forceinline__ void
k_loop(const unsigned short* __restrict__ xnv,
       const unsigned short* __restrict__ bp,
       int tbase, int row, int hi, int koTbase, v8f* acc) {
    for (int dt = 0; dt < KT; ++dt) {
        int t0 = tbase + row + dt - 4;        // temporal halo, pad=4
        int t1 = t0 + 16;
        bool va = !CHECK || (t0 >= 0 && t0 < TT);
        bool vb = !CHECK || (t1 >= 0 && t1 < TT);
        const uint4* arow0 = (const uint4*)(xnv + (long long)t0 * CCH + hi * 8);
        const uint4* arow1 = (const uint4*)(xnv + (long long)t1 * CCH + hi * 8);
        for (int cc = 0; cc < 8; ++cc) {
            Frag a0, a1;
            if (va) {                          // K = cc*32 + hi*8 +{0..7}, +16+{0..7}
                a0.q[0] = arow0[cc * 4];
                a0.q[1] = arow0[cc * 4 + 2];
            } else {
                a0.q[0] = make_uint4(0u, 0u, 0u, 0u);
                a0.q[1] = make_uint4(0u, 0u, 0u, 0u);
            }
            if (vb) {
                a1.q[0] = arow1[cc * 4];
                a1.q[1] = arow1[cc * 4 + 2];
            } else {
                a1.q[0] = make_uint4(0u, 0u, 0u, 0u);
                a1.q[1] = make_uint4(0u, 0u, 0u, 0u);
            }
            const uint4* bt = (const uint4*)(bp +
                (((long long)(dt * 8 + cc) * 48 + koTbase) * 512) + lane_chunk(row, hi));

            // Load all four B fragments first: 8 outstanding b128 loads in one
            // clause, drained with partial waits while earlier WMMAs execute.
            Frag b0, b1, b2, b3;
            b0.q[0] = bt[0];   b0.q[1] = bt[1];
            b1.q[0] = bt[64];  b1.q[1] = bt[65];
            b2.q[0] = bt[128]; b2.q[1] = bt[129];
            b3.q[0] = bt[192]; b3.q[1] = bt[193];

            acc[0] = __builtin_amdgcn_wmma_f32_16x16x32_bf16(
                false, a0.v, false, b0.v, (short)0, acc[0], false, false);
            acc[4] = __builtin_amdgcn_wmma_f32_16x16x32_bf16(
                false, a1.v, false, b0.v, (short)0, acc[4], false, false);
            acc[1] = __builtin_amdgcn_wmma_f32_16x16x32_bf16(
                false, a0.v, false, b1.v, (short)0, acc[1], false, false);
            acc[5] = __builtin_amdgcn_wmma_f32_16x16x32_bf16(
                false, a1.v, false, b1.v, (short)0, acc[5], false, false);
            acc[2] = __builtin_amdgcn_wmma_f32_16x16x32_bf16(
                false, a0.v, false, b2.v, (short)0, acc[2], false, false);
            acc[6] = __builtin_amdgcn_wmma_f32_16x16x32_bf16(
                false, a1.v, false, b2.v, (short)0, acc[6], false, false);
            acc[3] = __builtin_amdgcn_wmma_f32_16x16x32_bf16(
                false, a0.v, false, b3.v, (short)0, acc[3], false, false);
            acc[7] = __builtin_amdgcn_wmma_f32_16x16x32_bf16(
                false, a1.v, false, b3.v, (short)0, acc[7], false, false);
        }
    }
}

__global__ void __launch_bounds__(128)
st_gemm_kernel(const unsigned short* __restrict__ xb,   // [NV][T][C] bf16
               const unsigned short* __restrict__ bp,   // packed weights
               const float* __restrict__ bias,          // [KO]
               unsigned short* __restrict__ y)          // [NV][T][KO] bf16
{
    int b   = blockIdx.x;
    int koB = b % (KO / 64);                      // 12
    int tB  = (b / (KO / 64)) % (TT / 128);       // 4
    int nv  = b / ((KO / 64) * (TT / 128));       // 400
    int wave = threadIdx.x >> 5;
    int lane = threadIdx.x & 31;
    int row  = lane & 15;
    int hi   = lane >> 4;                         // K-half select
    int tbase   = tB * 128 + wave * 32;           // wave M-tile of 32 rows
    int koTbase = koB * 4;

    const unsigned short* xnv = xb + (long long)nv * TT * CCH;

    v8f acc[8];
#pragma unroll
    for (int j = 0; j < 8; ++j) acc[j] = v8f{};

    // interior waves never touch the t halo: wave-uniform branch, no exec churn
    bool interior = (tbase >= 4) && (tbase + 36 <= TT);
    if (interior) k_loop<false>(xnv, bp, tbase, row, hi, koTbase, acc);
    else          k_loop<true>(xnv, bp, tbase, row, hi, koTbase, acc);

    // Epilogue: C/D layout -> M = r + hi*8 (+16 for second A frag), N = lane&15
    long long ybase = (long long)nv * TT * KO;
#pragma unroll
    for (int m = 0; m < 2; ++m) {
#pragma unroll
        for (int j = 0; j < 4; ++j) {
            v8f a = acc[m * 4 + j];
            int ko = (koTbase + j) * 16 + row;
            float bv = bias[ko];
#pragma unroll
            for (int r = 0; r < 8; ++r) {
                int tout = tbase + m * 16 + hi * 8 + r;
                y[ybase + (long long)tout * KO + ko] = f2bf(a[r] + bv);
            }
        }
    }
}

// ---------------------------------------------------------------------------
// 4) Graph conv + ReLU: h[n][w][t][c] = relu(sum_{k,v} y[n][v][t][k*Co+c]*Ae[k][v][w])
//    FINAL=0 -> bf16 [n*V+w][t][c] (feeds next GEMM); FINAL=1 -> fp32 (N,T,V,C)
// ---------------------------------------------------------------------------
template <int FINAL>
__global__ void __launch_bounds__(256)
graph_conv_kernel(const unsigned short* __restrict__ y,  // [NV][T][KO] bf16
                  const float* __restrict__ A,
                  const float* __restrict__ ei,
                  unsigned short* __restrict__ hout,
                  float* __restrict__ fout) {
    __shared__ float Ae[KKG * VV * VV];
    for (int i = threadIdx.x; i < KKG * VV * VV; i += blockDim.x)
        Ae[i] = A[i] * ei[i];
    __syncthreads();

    long long g = (long long)blockIdx.x * blockDim.x + threadIdx.x; // N*T*C
    int c = (int)(g % CCH);
    int t = (int)((g / CCH) % TT);
    int n = (int)(g / ((long long)CCH * TT));

    float out[VV];
#pragma unroll
    for (int w = 0; w < VV; ++w) out[w] = 0.f;

    for (int k = 0; k < KKG; ++k) {
        for (int v = 0; v < VV; ++v) {
            float yv = bf2f(y[(((long long)(n * VV + v)) * TT + t) * KO + k * CO + c]);
            const float* arow = &Ae[(k * VV + v) * VV];
#pragma unroll
            for (int w = 0; w < VV; ++w) out[w] = fmaf(yv, arow[w], out[w]);
        }
    }

    if (FINAL) {
        for (int w = 0; w < VV; ++w)
            fout[(((long long)n * TT + t) * VV + w) * CCH + c] = fmaxf(out[w], 0.f);
    } else {
        for (int w = 0; w < VV; ++w)
            hout[(((long long)(n * VV + w)) * TT + t) * CCH + c] = f2bf(fmaxf(out[w], 0.f));
    }
}

// ---------------------------------------------------------------------------
extern "C" void kernel_launch(void* const* d_in, const int* in_sizes, int n_in,
                              void* d_out, int out_size, void* d_ws, size_t ws_size,
                              hipStream_t stream) {
    const float* x   = (const float*)d_in[0];
    const float* gam = (const float*)d_in[1];
    const float* bet = (const float*)d_in[2];
    const float* mu  = (const float*)d_in[3];
    const float* var = (const float*)d_in[4];
    const float* w1  = (const float*)d_in[5];
    const float* b1  = (const float*)d_in[6];
    const float* ei1 = (const float*)d_in[7];
    const float* w2  = (const float*)d_in[8];
    const float* b2  = (const float*)d_in[9];
    const float* ei2 = (const float*)d_in[10];
    const float* A   = (const float*)d_in[11];

    char* ws = (char*)d_ws;
    const long long XB_BYTES = (long long)NB * VV * TT * CCH * 2;  // 104,857,600
    const long long Y_BYTES  = (long long)NB * VV * TT * KO * 2;   // 314,572,800
    const long long BP_ELEMS = (long long)KT * 8 * 48 * 512;       // 1,769,472

    unsigned short* xb   = (unsigned short*)ws;                    // also reused as h
    unsigned short* ybuf = (unsigned short*)(ws + XB_BYTES);
    unsigned short* bp1  = (unsigned short*)(ws + XB_BYTES + Y_BYTES);
    unsigned short* bp2  = bp1 + BP_ELEMS;

    const int GEMM_BLOCKS = NB * VV * (TT / 128) * (KO / 64);      // 19,200
    const int GC_BLOCKS   = (NB * TT * CCH) / 256;                 // 8,192

    bn_pack_kernel<<<(NB * TT * VV * CCH) / 256, 256, 0, stream>>>(x, gam, bet, mu, var, xb);
    pack_w_kernel<<<(int)(BP_ELEMS / 256), 256, 0, stream>>>(w1, bp1);
    pack_w_kernel<<<(int)(BP_ELEMS / 256), 256, 0, stream>>>(w2, bp2);

    st_gemm_kernel<<<GEMM_BLOCKS, 128, 0, stream>>>(xb, bp1, b1, ybuf);
    graph_conv_kernel<0><<<GC_BLOCKS, 256, 0, stream>>>(ybuf, A, ei1, xb, nullptr);
    st_gemm_kernel<<<GEMM_BLOCKS, 128, 0, stream>>>(xb, bp2, b2, ybuf);
    graph_conv_kernel<1><<<GC_BLOCKS, 256, 0, stream>>>(ybuf, A, ei2, nullptr, (float*)d_out);
}
